// TestTimeAttention_18829136626262
// MI455X (gfx1250) — matrix operand order repaired and checked by reading, not compiled, not run on hardware
//
#include <hip/hip_runtime.h>
#include <hip/hip_bf16.h>
#include <stdint.h>
#include <stddef.h>

// Problem dims (fixed by the reference)
#define BDIM 64   // batch
#define NNGH 16   // neighbors
#define TT   64   // time steps
#define DD   128  // input dim
#define HH   256  // hidden dim
#define GG   768  // 3*H (gates)
#define MMSL 17   // N+1 slots

typedef __attribute__((ext_vector_type(16))) __bf16        v16bf;
typedef __attribute__((ext_vector_type(8)))  float         v8f;
typedef __attribute__((ext_vector_type(4)))  float         v4f;
typedef __attribute__((ext_vector_type(8)))  unsigned int  v8u;
typedef __attribute__((ext_vector_type(4)))  unsigned int  v4u;
typedef __attribute__((ext_vector_type(8)))  int           v8i;
typedef __attribute__((ext_vector_type(4)))  int           v4i;

#if defined(__HIP_DEVICE_COMPILE__) && __has_builtin(__builtin_amdgcn_tensor_load_to_lds)
#define USE_TDM 1
#else
#define USE_TDM 0
#endif

// ---------------------------------------------------------------------------
// WMMA helpers (CDNA5 wave32, V_WMMA_F32_16X16X32_BF16)
// ---------------------------------------------------------------------------
static __device__ __forceinline__ v8f wmma_bf16(v16bf a, v16bf b, v8f c) {
  return __builtin_amdgcn_wmma_f32_16x16x32_bf16(false, a, false, b, (short)0, c,
                                                 false, false);
}

// A fragment 16x32 (MxK) from fp32 memory (p must be 32B aligned).
// lane row m = lane&15; elems 0..15 -> K = hi*16+j.
static __device__ __forceinline__ v16bf frag_a_f32(const float* p, int hi) {
  const v4f* q = (const v4f*)(p + hi * 16);
  v4f a0 = q[0], a1 = q[1], a2 = q[2], a3 = q[3];
  v16bf a;
#pragma unroll
  for (int j = 0; j < 4; ++j) {
    a[j]      = (__bf16)a0[j];
    a[j + 4]  = (__bf16)a1[j];
    a[j + 8]  = (__bf16)a2[j];
    a[j + 12] = (__bf16)a3[j];
  }
  return a;
}

// B fragment 32x16 (KxN), B = W^T with W (N x K) bf16 row-major; contiguous
// 32-byte load (p = W + n*ldw + k0 + hi*16 must be 32B aligned).
static __device__ __forceinline__ v16bf frag_b_tr_bf16(const __bf16* p) {
  v8u u = *(const v8u*)p;
  return __builtin_bit_cast(v16bf, u);
}

// ---------------------------------------------------------------------------
// Tensor Data Mover: async DMA of one 16-row x 128-f32 tile (row stride T*D)
// from global memory into LDS (ISA 08_async_tensor, D# groups per section 8).
// This toolchain exposes the 6-arg builtin:
//   (uint32x4 g0, int32x8 g1, int32x4 g2, int32x4 g3, int32x8 pad, i32 cpol)
// ---------------------------------------------------------------------------
#if USE_TDM
static __device__ __forceinline__ void tdm_load_x16(const float* gptr,
                                                    unsigned lds_byte) {
  unsigned long long ga = (unsigned long long)(size_t)gptr;
  v4u g0;
  g0[0] = 1u;                                   // count=1, user descriptor
  g0[1] = lds_byte;                             // LDS byte address of tile
  g0[2] = (unsigned)(ga & 0xffffffffu);         // global_addr[31:0]
  g0[3] = (unsigned)((ga >> 32) & 0x01ffffffu)  // global_addr[56:32]
          | (2u << 30);                         // type = 2 ("image")
  v8i g1;
  g1[0] = 0x00020000;                           // data_size = 4 bytes
  g1[1] = (int)(DD << 16);                      // tensor_dim0 = 128
  g1[2] = (int)(16 << 16);                      // tensor_dim1 = 16
  g1[3] = (int)(DD << 16);                      // tile_dim0 = 128
  g1[4] = 16;                                   // tile_dim1 = 16
  g1[5] = (int)(TT * DD);                       // tensor_dim0_stride = 8192
  g1[6] = 0;
  g1[7] = 0;
  v4i z4 = {0, 0, 0, 0};                        // groups 2/3 unused (2-D)
  v8i z8 = {0, 0, 0, 0, 0, 0, 0, 0};
  __builtin_amdgcn_tensor_load_to_lds(g0, g1, z4, z4, z8, 0);
}
#endif

// ---------------------------------------------------------------------------
// fp32 -> bf16 conversion (optionally transposed)
// ---------------------------------------------------------------------------
__global__ void f32_to_bf16_kernel(const float* __restrict__ s,
                                   __bf16* __restrict__ d, int n) {
  int i = blockIdx.x * blockDim.x + threadIdx.x;
  if (i < n) d[i] = (__bf16)s[i];
}

// s is (K x N) row-major; d becomes (N x K) row-major (so d acts as W^T).
__global__ void f32_to_bf16_tr_kernel(const float* __restrict__ s,
                                      __bf16* __restrict__ d, int K, int N) {
  int i = blockIdx.x * blockDim.x + threadIdx.x;
  if (i < K * N) {
    int k = i / N, n = i % N;
    d[(size_t)n * K + k] = (__bf16)s[i];
  }
}

// ---------------------------------------------------------------------------
// WMMA GEMM: C(M x 256) = A(M x 256, f32) @ W^T ; W (256 x 256) bf16.
// One wave per 16x16 tile. grid = (16, M/16), block = 32.
// ---------------------------------------------------------------------------
__global__ __launch_bounds__(32)
void wmma_gemm_kernel(const float* __restrict__ A,
                      const __bf16* __restrict__ W,
                      float* __restrict__ C) {
  const int lane = threadIdx.x & 31;
  const int hi = lane >> 4, mn = lane & 15;
  const float* arow = A + (size_t)(blockIdx.y * 16 + mn) * HH;
  const int col = blockIdx.x * 16 + mn;
  const __bf16* wrow = W + (size_t)col * HH + hi * 16;
  v8f acc = {};
#pragma unroll
  for (int k0 = 0; k0 < HH; k0 += 32) {
    __builtin_prefetch(arow + k0 + 64, 0, 1);  // global_prefetch_b8
    acc = wmma_bf16(frag_a_f32(arow + k0, hi), frag_b_tr_bf16(wrow + k0), acc);
  }
#pragma unroll
  for (int r = 0; r < 8; ++r)
    C[(size_t)(blockIdx.y * 16 + r + hi * 8) * HH + col] = acc[r];
}

// ---------------------------------------------------------------------------
// GRU scan. Each workgroup (8 waves) owns 16 independent sequences; hidden
// state lives in LDS across all T steps. Per step both gate GEMMs run on
// WMMA with hoisted A-fragments; x_{t+1} is DMA'd by the Tensor Data Mover
// into a double-buffered LDS staging tile, overlapped with step-t compute.
// Dynamic LDS: h(16x256) + gi(16x768) + gh(16x768) + xstage(2x16x128) = 128KB.
// ---------------------------------------------------------------------------
__global__ __launch_bounds__(256)
void gru_scan_kernel(const float* __restrict__ x,    // (rows, T, D)
                     const float* __restrict__ h0,   // (rows, H)
                     const __bf16* __restrict__ Wih, // (3H, D) bf16
                     const __bf16* __restrict__ Whh, // (3H, H) bf16
                     const float* __restrict__ bih,
                     const float* __restrict__ bhh,
                     float* __restrict__ ys,         // (rows, T, H)
                     float* __restrict__ hT) {       // (rows, H)
  extern __shared__ char smem_scan[];
  float* hS  = (float*)smem_scan;   // 16 x 256
  float* giS = hS + 16 * HH;        // 16 x 768
  float* ghS = giS + 16 * GG;       // 16 x 768
  float* xS  = ghS + 16 * GG;       // 2 x 16 x 128 staging

  const int tid = threadIdx.x;
  const int wave = tid >> 5, lane = tid & 31;
  const int hi = lane >> 4, mn = lane & 15;
  const int rowBase = blockIdx.x * 16;

#pragma unroll
  for (int e = 0; e < 16; ++e) {
    int f = tid + 256 * e;  // 0..4095
    hS[f] = h0[(size_t)(rowBase + (f >> 8)) * HH + (f & 255)];
  }
#if USE_TDM
  const unsigned xsOff = (unsigned)(size_t)(const void*)xS;  // LDS byte addr
  if (wave == 0) {
    tdm_load_x16(x + (size_t)rowBase * TT * DD, xsOff);      // preload t=0
    __builtin_amdgcn_s_wait_tensorcnt(0);
  }
#endif
  __syncthreads();

  for (int t = 0; t < TT; ++t) {
#if USE_TDM
    if (wave == 0 && t + 1 < TT)                             // prefetch t+1
      tdm_load_x16(x + ((size_t)rowBase * TT + (t + 1)) * DD,
                   xsOff + (unsigned)(((t + 1) & 1) * 16 * DD * 4));
    const float* xrow = xS + (t & 1) * 16 * DD + mn * DD;
#else
    const float* xrow = x + ((size_t)(rowBase + mn) * TT + t) * DD;
#endif
    const float* hrow = hS + mn * HH;

    // Hoist all A-fragments for this step (same rows for every column tile).
    v16bf xa[DD / 32], ha[HH / 32];
#pragma unroll
    for (int k = 0; k < DD / 32; ++k) xa[k] = frag_a_f32(xrow + k * 32, hi);
#pragma unroll
    for (int k = 0; k < HH / 32; ++k) ha[k] = frag_a_f32(hrow + k * 32, hi);

    // 48 column tiles over 8 waves -> 6 per wave.
#pragma unroll
    for (int i = 0; i < 6; ++i) {
      const int col = (wave + 8 * i) * 16 + mn;  // 0..767
      const __bf16* wi = Wih + (size_t)col * DD + hi * 16;
      const __bf16* wh = Whh + (size_t)col * HH + hi * 16;
      v8f ai = {}, ah = {};
#pragma unroll
      for (int k = 0; k < DD / 32; ++k)
        ai = wmma_bf16(xa[k], frag_b_tr_bf16(wi + k * 32), ai);
#pragma unroll
      for (int k = 0; k < HH / 32; ++k)
        ah = wmma_bf16(ha[k], frag_b_tr_bf16(wh + k * 32), ah);
      const float bi = bih[col], bh = bhh[col];
#pragma unroll
      for (int r = 0; r < 8; ++r) {
        const int m = r + hi * 8;
        giS[m * GG + col] = ai[r] + bi;
        ghS[m * GG + col] = ah[r] + bh;
      }
    }
    __syncthreads();

    // Elementwise gates + state update (4096 elems over 256 threads).
#pragma unroll
    for (int e = 0; e < 16; ++e) {
      int f = tid + 256 * e;
      int m = f >> 8, j = f & 255;
      float gr = giS[m * GG + j]       + ghS[m * GG + j];
      float gz = giS[m * GG + 256 + j] + ghS[m * GG + 256 + j];
      float r  = 1.0f / (1.0f + __expf(-gr));
      float z  = 1.0f / (1.0f + __expf(-gz));
      float n  = tanhf(giS[m * GG + 512 + j] + r * ghS[m * GG + 512 + j]);
      float hn = (1.0f - z) * n + z * hS[m * HH + j];
      hS[m * HH + j] = hn;
      ys[((size_t)(rowBase + m) * TT + t) * HH + j] = hn;
    }
#if USE_TDM
    if (wave == 0) __builtin_amdgcn_s_wait_tensorcnt(0);  // drain t+1 DMA
#endif
    __syncthreads();
  }

#pragma unroll
  for (int e = 0; e < 16; ++e) {
    int f = tid + 256 * e;
    hT[(size_t)(rowBase + (f >> 8)) * HH + (f & 255)] = hS[f];
  }
}

// ---------------------------------------------------------------------------
// Fused attention per (b, m):
//   K = kv @ Wk^T (LDS, row-major) ; V = kv @ Wv^T (LDS, TRANSPOSED)
//   S = Qb_b @ K^T -> causal softmax -> A (straight to d_out)
//   ctx = A @ V ; ctxdot[b,m,t] = ctx[t,:] . wprj  (O=1 folded into LDS)
// A-fragments hoisted per wave (row-tile rt = wave>>1 is wave-invariant).
// Dynamic LDS: Ks 32KB bf16 + Vt 32KB bf16 + Ss 16KB f32 + oac 256B.
// ---------------------------------------------------------------------------
__global__ __launch_bounds__(256)
void attention_kernel(const float* __restrict__ Qb,       // (B*T, H)
                      const float* __restrict__ node_out, // (B, T, H)
                      const float* __restrict__ ngh_out,  // (B*N, T, H)
                      const __bf16* __restrict__ Wk,      // (H, H)
                      const __bf16* __restrict__ Wv,      // (H, H)
                      const float* __restrict__ bbil,     // (T)
                      const float* __restrict__ Wprj,     // (H)
                      float* __restrict__ A_out,          // (B, M, T, T)
                      float* __restrict__ ctxdot) {       // (B, M, T)
  extern __shared__ char smem_att[];
  __bf16* Ks = (__bf16*)smem_att;        // (T x H): K row-major
  __bf16* Vt = Ks + TT * HH;             // (H x T): V transposed
  float*  Ss = (float*)(Vt + HH * TT);   // (T x T)
  float* oac = Ss + TT * TT;             // (T)

  const int tid = threadIdx.x;
  const int wave = tid >> 5, lane = tid & 31;
  const int hi = lane >> 4, mn = lane & 15;
  const int b = blockIdx.x / MMSL, m = blockIdx.x % MMSL;
  const int rt = wave >> 1;              // fixed row tile per wave

  const float* kvbase = (m == 0)
      ? node_out + (size_t)b * TT * HH
      : ngh_out + ((size_t)b * NNGH + (m - 1)) * TT * HH;

  // --- K/V projection: hoist kv A-fragments, 8 column tiles per wave ---
  {
    const float* arow = kvbase + (size_t)(rt * 16 + mn) * HH;
    v16bf av[HH / 32];
#pragma unroll
    for (int k = 0; k < HH / 32; ++k) av[k] = frag_a_f32(arow + k * 32, hi);
#pragma unroll
    for (int i = 0; i < 8; ++i) {
      const int ct = (wave & 1) * 8 + i;
      const int col = ct * 16 + mn;
      const __bf16* wk = Wk + (size_t)col * HH + hi * 16;
      const __bf16* wv = Wv + (size_t)col * HH + hi * 16;
      v8f aK = {}, aV = {};
#pragma unroll
      for (int k = 0; k < HH / 32; ++k) {
        aK = wmma_bf16(av[k], frag_b_tr_bf16(wk + k * 32), aK);
        aV = wmma_bf16(av[k], frag_b_tr_bf16(wv + k * 32), aV);
      }
#pragma unroll
      for (int r = 0; r < 8; ++r) {
        const int srow = rt * 16 + r + hi * 8;
        Ks[srow * HH + col] = (__bf16)aK[r];
        Vt[col * TT + srow] = (__bf16)aV[r];   // store V transposed
      }
    }
  }
  __syncthreads();

  // --- S = Qb_b (64x256) @ K^T : 2 tiles per wave, hoisted Q-fragments ---
  {
    const float* qrow = Qb + ((size_t)b * TT + rt * 16 + mn) * HH;
    v16bf qa[HH / 32];
#pragma unroll
    for (int k = 0; k < HH / 32; ++k) qa[k] = frag_a_f32(qrow + k * 32, hi);
#pragma unroll
    for (int i = 0; i < 2; ++i) {
      const int ct = (wave & 1) * 2 + i;
      const __bf16* krow = Ks + (size_t)(ct * 16 + mn) * HH + hi * 16;
      v8f acc = {};
#pragma unroll
      for (int k = 0; k < HH / 32; ++k)
        acc = wmma_bf16(qa[k], frag_b_tr_bf16(krow + k * 32), acc);
#pragma unroll
      for (int r = 0; r < 8; ++r)
        Ss[(rt * 16 + r + hi * 8) * TT + ct * 16 + mn] = acc[r];
    }
  }
  __syncthreads();

  // --- causal softmax per row (threads 0..63); A goes straight to d_out ---
  if (tid < TT) {
    const int t = tid;
    float* row = Ss + t * TT;
    float mx = -3.4e38f;
    for (int s = 0; s <= t; ++s) { float v = row[s] + bbil[s]; mx = fmaxf(mx, v); }
    float sum = 0.0f;
    for (int s = 0; s < TT; ++s) {
      float a = (s <= t) ? __expf(row[s] + bbil[s] - mx) : 0.0f;
      row[s] = a; sum += a;
    }
    const float inv = 1.0f / sum;
    float* arow = A_out + (((size_t)b * MMSL + m) * TT + t) * TT;
    for (int s = 0; s < TT; ++s) { float a = row[s] * inv; row[s] = a; arow[s] = a; }
    oac[t] = 0.0f;
  }
  __syncthreads();

  // --- ctx = A @ V via V^T (contiguous B-fragments), O=1 LDS reduction ---
  {
    const float* srow = Ss + (size_t)(rt * 16 + mn) * TT;
    v16bf sa[TT / 32];
#pragma unroll
    for (int k = 0; k < TT / 32; ++k) sa[k] = frag_a_f32(srow + k * 32, hi);
#pragma unroll
    for (int i = 0; i < 8; ++i) {
      const int ct = (wave & 1) * 8 + i;
      const int col = ct * 16 + mn;
      const __bf16* vrow = Vt + (size_t)col * TT + hi * 16;
      v8f acc = {};
#pragma unroll
      for (int k = 0; k < TT / 32; ++k)
        acc = wmma_bf16(sa[k], frag_b_tr_bf16(vrow + k * 32), acc);
      const float w = Wprj[col];
#pragma unroll
      for (int r = 0; r < 8; ++r)
        atomicAdd(&oac[rt * 16 + r + hi * 8], acc[r] * w);
    }
  }
  __syncthreads();
  if (tid < TT)
    ctxdot[((size_t)b * MMSL + m) * TT + tid] = oac[tid];
}

// out[b,t] = M*bprj + sum_m ctxdot[b,m,t]
__global__ void reduce_out_kernel(const float* __restrict__ ctxdot,
                                  const float* __restrict__ bprj,
                                  float* __restrict__ out) {
  int i = blockIdx.x * blockDim.x + threadIdx.x;
  if (i < BDIM * TT) {
    int b = i >> 6, t = i & 63;
    float s = (float)MMSL * bprj[0];
    for (int m = 0; m < MMSL; ++m) s += ctxdot[((size_t)b * MMSL + m) * TT + t];
    out[i] = s;
  }
}

// ---------------------------------------------------------------------------
extern "C" void kernel_launch(void* const* d_in, const int* in_sizes, int n_in,
                              void* d_out, int out_size, void* d_ws, size_t ws_size,
                              hipStream_t stream) {
  (void)in_sizes; (void)n_in; (void)out_size; (void)ws_size;

  const float* node_input = (const float*)d_in[0];   // (B,T,D)
  const float* node_hidden = (const float*)d_in[1];  // (1,B,H)
  const float* ngh_input = (const float*)d_in[2];    // (B,N,T,D)
  const float* ngh_hidden = (const float*)d_in[3];   // (1,B*N,H)
  /* d_in[4] = s_len : unused by the reference body */
  const float* Wih_node = (const float*)d_in[5];
  const float* Whh_node = (const float*)d_in[6];
  const float* bih_node = (const float*)d_in[7];
  const float* bhh_node = (const float*)d_in[8];
  const float* Wih_ngh = (const float*)d_in[9];
  const float* Whh_ngh = (const float*)d_in[10];
  const float* bih_ngh = (const float*)d_in[11];
  const float* bhh_ngh = (const float*)d_in[12];
  const float* Wq   = (const float*)d_in[13];
  const float* Wk   = (const float*)d_in[14];
  const float* Wv   = (const float*)d_in[15];
  const float* Wbil = (const float*)d_in[16];
  const float* bbil = (const float*)d_in[17];
  const float* Wprj = (const float*)d_in[18];
  const float* bprj = (const float*)d_in[19];

  // d_out layout: out(4096) | node_hT(16384) | ngh_hT(262144) | A(4456448)
  float* out_f   = (float*)d_out;
  float* out_sum = out_f;
  float* node_hT = out_f + 4096;
  float* ngh_hT  = out_f + 4096 + 16384;
  float* A_out   = out_f + 4096 + 16384 + 262144;

  // Workspace partition (~81 MB total).
  char* ws = (char*)d_ws;
  auto alloc = [&](size_t bytes) {
    char* p = ws;
    ws += (bytes + 255) & ~(size_t)255;
    return p;
  };
  float* node_out = (float*)alloc((size_t)BDIM * TT * HH * 4);         // 4 MB
  float* ngh_out  = (float*)alloc((size_t)BDIM * NNGH * TT * HH * 4);  // 64 MB
  float* Qbuf     = (float*)alloc((size_t)BDIM * TT * HH * 4);         // 4 MB
  float* Qb       = (float*)alloc((size_t)BDIM * TT * HH * 4);         // 4 MB
  float* ctxdot   = (float*)alloc((size_t)BDIM * MMSL * TT * 4);       // 278 KB
  __bf16* Wih_node_b = (__bf16*)alloc((size_t)GG * DD * 2);
  __bf16* Whh_node_b = (__bf16*)alloc((size_t)GG * HH * 2);
  __bf16* Wih_ngh_b  = (__bf16*)alloc((size_t)GG * DD * 2);
  __bf16* Whh_ngh_b  = (__bf16*)alloc((size_t)GG * HH * 2);
  __bf16* Wq_b    = (__bf16*)alloc((size_t)HH * HH * 2);
  __bf16* Wk_b    = (__bf16*)alloc((size_t)HH * HH * 2);
  __bf16* Wv_b    = (__bf16*)alloc((size_t)HH * HH * 2);
  __bf16* WbilT_b = (__bf16*)alloc((size_t)HH * HH * 2);  // transposed

  auto cvt = [&](const float* s, __bf16* d, int n) {
    f32_to_bf16_kernel<<<(n + 255) / 256, 256, 0, stream>>>(s, d, n);
  };
  cvt(Wih_node, Wih_node_b, GG * DD);
  cvt(Whh_node, Whh_node_b, GG * HH);
  cvt(Wih_ngh,  Wih_ngh_b,  GG * DD);
  cvt(Whh_ngh,  Whh_ngh_b,  GG * HH);
  cvt(Wq, Wq_b, HH * HH);
  cvt(Wk, Wk_b, HH * HH);
  cvt(Wv, Wv_b, HH * HH);
  f32_to_bf16_tr_kernel<<<(HH * HH + 255) / 256, 256, 0, stream>>>(Wbil, WbilT_b,
                                                                   HH, HH);

  // GRU scans (node: 4 row-blocks, neighbors: 64 row-blocks).
  const int scanShm = (16 * HH + 2 * 16 * GG + 2 * 16 * DD) * 4;  // 128 KB
  (void)hipFuncSetAttribute(reinterpret_cast<const void*>(gru_scan_kernel),
                            hipFuncAttributeMaxDynamicSharedMemorySize, scanShm);
  gru_scan_kernel<<<BDIM / 16, 256, scanShm, stream>>>(
      node_input, node_hidden, Wih_node_b, Whh_node_b, bih_node, bhh_node,
      node_out, node_hT);
  gru_scan_kernel<<<(BDIM * NNGH) / 16, 256, scanShm, stream>>>(
      ngh_input, ngh_hidden, Wih_ngh_b, Whh_ngh_b, bih_ngh, bhh_ngh,
      ngh_out, ngh_hT);

  // Q = node_out @ Wq^T ; Qb = Q @ Wbil (via pre-transposed WbilT)
  dim3 ggrid(HH / 16, (BDIM * TT) / 16);  // (16, 256)
  wmma_gemm_kernel<<<ggrid, 32, 0, stream>>>(node_out, Wq_b, Qbuf);
  wmma_gemm_kernel<<<ggrid, 32, 0, stream>>>(Qbuf, WbilT_b, Qb);

  // Fused attention per (b, m).
  const int attShm = TT * HH * 2 * 2 + TT * TT * 4 + TT * 4;  // ~80.3 KB
  (void)hipFuncSetAttribute(reinterpret_cast<const void*>(attention_kernel),
                            hipFuncAttributeMaxDynamicSharedMemorySize, attShm);
  attention_kernel<<<BDIM * MMSL, 256, attShm, stream>>>(
      Qb, node_out, ngh_out, Wk_b, Wv_b, bbil, Wprj, A_out, ctxdot);

  // Final slot-sum + bias.
  reduce_out_kernel<<<(BDIM * TT + 255) / 256, 256, 0, stream>>>(ctxdot, bprj,
                                                                 out_sum);
}
